// NonLocal_71923522339105
// MI455X (gfx1250) — matrix operand order repaired and checked by reading, not compiled, not run on hardware
//
#include <hip/hip_runtime.h>
#include <hip/hip_bf16.h>
#include <math.h>

#define BB  4
#define CC  512
#define HH  4096
#define ICC 256

typedef __bf16 bf16_t;
typedef bf16_t v16bf __attribute__((ext_vector_type(16)));
typedef float  v8f   __attribute__((ext_vector_type(8)));
typedef unsigned int v4u __attribute__((ext_vector_type(4)));
typedef int v4i __attribute__((ext_vector_type(4)));
typedef int v8i __attribute__((ext_vector_type(8)));

// A-fragment (16x32, 16-bit) K index for element j given lane-half kh:
// lane<16: j=0..7 -> K0..7, j=8..15 -> K16..23 ; lane>=16: +8
__device__ __forceinline__ int a_k(int j, int kh) {
    return ((j & 8) << 1) + kh * 8 + (j & 7);
}

// ---------------------------------------------------------------------------
// TDM: 2-D tile load Global -> LDS via tensor descriptor (D#).
// Descriptor layout per CDNA5 ISA (data_size = 2 bytes).
// 6-arg builtin form (clang-23 / therock-10.0 toolchain).
// ---------------------------------------------------------------------------
__device__ __forceinline__ void tdm_load_2d(unsigned int lds_byte_addr,
                                            const void* gaddr,
                                            unsigned int tile0, unsigned int tile1,
                                            unsigned long long stride0,
                                            unsigned int dim0, unsigned int dim1)
{
    unsigned long long ga = (unsigned long long)(uintptr_t)gaddr;
    v4u g0;
    g0[0] = 1u;                                   // count=1, user descriptor
    g0[1] = lds_byte_addr;                        // lds_addr (bytes)
    g0[2] = (unsigned int)ga;                     // global_addr[31:0]
    g0[3] = ((unsigned int)(ga >> 32) & 0x01FFFFFFu) | (2u << 30); // addr[56:32], type=2

    v8i g1;
    g1[0] = (int)(1u << 16);                      // workgroup_mask=0, data_size=1 (2B)
    g1[1] = (int)((dim0 & 0xFFFFu) << 16);        // tensor_dim0[15:0]
    g1[2] = (int)(((dim0 >> 16) & 0xFFFFu) | ((dim1 & 0xFFFFu) << 16));
    g1[3] = (int)(((dim1 >> 16) & 0xFFFFu) | (tile0 << 16));   // tile_dim0
    g1[4] = (int)(tile1 & 0xFFFFu);               // tile_dim1, tile_dim2=0
    g1[5] = (int)(unsigned int)(stride0 & 0xFFFFFFFFu);        // tensor_dim0_stride
    g1[6] = (int)(unsigned int)((stride0 >> 32) & 0xFFFFu);    // stride0 hi, stride1=0
    g1[7] = 0;

    v4i z4 = {};
    v8i z8 = {};
    __builtin_amdgcn_tensor_load_to_lds(g0, g1, z4, z4, z8, 0);
}

// -------------------------------------------------------------------------
// K1: projections. phi -> [b][i][h]; theta,g -> [b][h][i] (pre-transposed)
// Wave tile: M16 x N64. Block: 8 waves = M32 x N256.
// -------------------------------------------------------------------------
__global__ __launch_bounds__(256) void proj_kernel(
    const float* __restrict__ x,
    const float* __restrict__ w_phi,   const float* __restrict__ b_phi,
    const float* __restrict__ w_theta, const float* __restrict__ b_theta,
    const float* __restrict__ w_g,     const float* __restrict__ b_g,
    bf16_t* __restrict__ phiB, bf16_t* __restrict__ thetaB, bf16_t* __restrict__ gB)
{
    const int b = blockIdx.z / 3, proj = blockIdx.z % 3;
    const int w = threadIdx.x >> 5, lane = threadIdx.x & 31;
    const int wm = w >> 2, wn = w & 3;
    const int l0 = lane & 15, kh = lane >> 4;
    const int ibase = blockIdx.y * 32 + wm * 16;
    const int hbase = blockIdx.x * 256 + wn * 64;

    const float* W; const float* bias;
    if (proj == 0)      { W = w_phi;   bias = b_phi;   }
    else if (proj == 1) { W = w_theta; bias = b_theta; }
    else                { W = w_g;     bias = b_g;     }

    const float* xb = x + (size_t)b * CC * HH;

    v8f acc[4] = {};
    for (int kb = 0; kb < CC; kb += 32) {
        v16bf a;
        const float* wrow = W + (size_t)(ibase + l0) * CC + kb;
        #pragma unroll
        for (int j = 0; j < 16; ++j) a[j] = (bf16_t)wrow[kh * 16 + j];
        #pragma unroll
        for (int t = 0; t < 4; ++t) {
            v16bf bfrag;
            const float* xcol = xb + (size_t)(kb + kh * 16) * HH + hbase + t * 16 + l0;
            #pragma unroll
            for (int j = 0; j < 16; ++j) bfrag[j] = (bf16_t)xcol[(size_t)j * HH];
            acc[t] = __builtin_amdgcn_wmma_f32_16x16x32_bf16(
                false, a, false, bfrag, (short)0, acc[t], false, false);
        }
    }
    #pragma unroll
    for (int t = 0; t < 4; ++t) {
        #pragma unroll
        for (int r = 0; r < 8; ++r) {
            const int i = ibase + kh * 8 + r;
            const int h = hbase + t * 16 + l0;
            const float v = acc[t][r] + bias[i];
            if (proj == 0)
                phiB[((size_t)b * ICC + i) * HH + h] = (bf16_t)v;
            else if (proj == 1)
                thetaB[((size_t)b * HH + h) * ICC + i] = (bf16_t)v;
            else
                gB[((size_t)b * HH + h) * ICC + i] = (bf16_t)v;
        }
    }
}

// -------------------------------------------------------------------------
// K2: scores[q][k] = sum_i theta[q][i]*phi[i][k]  (M=N=4096, K=256), bf16 out
// TDM stages theta (64x32) + phi (32x128) tiles into LDS each K-step.
// Wave tile: M32 x N32 (4 accums). Block: 8 waves = M64 x N128.
// -------------------------------------------------------------------------
__global__ __launch_bounds__(256) void scores_kernel(
    const bf16_t* __restrict__ thetaB, const bf16_t* __restrict__ phiB,
    bf16_t* __restrict__ scoresB)
{
    __shared__ bf16_t sTheta[64 * 32];   // [q][i], row stride 32
    __shared__ bf16_t sPhi[32 * 128];    // [i][k], row stride 128

    const int b = blockIdx.z;
    const int w = threadIdx.x >> 5, lane = threadIdx.x & 31;
    const int wm = w >> 2, wn = w & 3;
    const int l0 = lane & 15, kh = lane >> 4;
    const int qblk = blockIdx.y * 64;
    const int kblk = blockIdx.x * 128;

    const bf16_t* th = thetaB + (size_t)b * HH * ICC + (size_t)qblk * ICC;
    const bf16_t* ph = phiB   + (size_t)b * ICC * HH + kblk;

    const unsigned int ldsTheta = (unsigned int)(uintptr_t)&sTheta[0];
    const unsigned int ldsPhi   = (unsigned int)(uintptr_t)&sPhi[0];

    v8f acc[2][2] = {};
    for (int ib = 0; ib < ICC; ib += 32) {
        __syncthreads();                      // previous iteration's LDS reads done
        if (w == 0) {
            // theta tile: 64 rows (q) x 32 cols (i), row stride ICC
            tdm_load_2d(ldsTheta, (const void*)(th + ib), 32, 64, ICC, ICC, HH);
            // phi tile: 32 rows (i) x 128 cols (k), row stride HH
            tdm_load_2d(ldsPhi, (const void*)(ph + (size_t)ib * HH), 128, 32, HH, HH, ICC);
            __builtin_amdgcn_s_wait_tensorcnt(0);
        }
        __syncthreads();                      // tiles visible to all waves

        v16bf a[2], bfrag[2];
        #pragma unroll
        for (int m = 0; m < 2; ++m) {
            // A layout: per lane 16 contiguous K (k = 16*kh + j) -> two 16B LDS loads
            const bf16_t* arow = &sTheta[(wm * 32 + m * 16 + l0) * 32];
            union { v16bf v; v4u u[2]; } fa;
            fa.u[0] = *(const v4u*)(arow + kh * 16);
            fa.u[1] = *(const v4u*)(arow + kh * 16 + 8);
            a[m] = fa.v;
        }
        #pragma unroll
        for (int n = 0; n < 2; ++n) {
            const bf16_t* bcol = &sPhi[(kh * 16) * 128 + wn * 32 + n * 16 + l0];
            #pragma unroll
            for (int j = 0; j < 16; ++j) bfrag[n][j] = bcol[j * 128];
        }
        #pragma unroll
        for (int m = 0; m < 2; ++m)
            #pragma unroll
            for (int n = 0; n < 2; ++n)
                acc[m][n] = __builtin_amdgcn_wmma_f32_16x16x32_bf16(
                    false, a[m], false, bfrag[n], (short)0, acc[m][n], false, false);
    }

    bf16_t* sb = scoresB + (size_t)b * HH * HH;
    #pragma unroll
    for (int m = 0; m < 2; ++m)
        #pragma unroll
        for (int n = 0; n < 2; ++n)
            #pragma unroll
            for (int r = 0; r < 8; ++r) {
                const int q = qblk + wm * 32 + m * 16 + kh * 8 + r;
                const int k = kblk + wn * 32 + n * 16 + l0;
                sb[(size_t)q * HH + k] = (bf16_t)acc[m][n][r];
            }
}

// -------------------------------------------------------------------------
// K3: column-wise (axis=q) softmax stats: colmax[k], colinv[k]=1/sum exp
// -------------------------------------------------------------------------
__global__ __launch_bounds__(256) void colsoftmax_kernel(
    const bf16_t* __restrict__ scoresB,
    float* __restrict__ colmax, float* __restrict__ colinv)
{
    const int b = blockIdx.y;
    const int k = blockIdx.x * 256 + threadIdx.x;
    const bf16_t* col = scoresB + (size_t)b * HH * HH + k;
    float m = -1e30f;
    for (int q = 0; q < HH; ++q) m = fmaxf(m, (float)col[(size_t)q * HH]);
    float s = 0.f;
    for (int q = 0; q < HH; ++q) s += __expf((float)col[(size_t)q * HH] - m);
    colmax[b * HH + k] = m;
    colinv[b * HH + k] = 1.0f / s;
}

// -------------------------------------------------------------------------
// K4: ag[q][i] = sum_k softmax_col(scores)[q][k] * g[k][i]
//     (M=4096, N=256, K=4096). attn computed on the fly in A-fragment.
// Wave tile: M16 x N64. Block: 8 waves = M64 x N128.
// -------------------------------------------------------------------------
__global__ __launch_bounds__(256) void ag_kernel(
    const bf16_t* __restrict__ scoresB,
    const float* __restrict__ colmax, const float* __restrict__ colinv,
    const bf16_t* __restrict__ gB, bf16_t* __restrict__ agB)
{
    const int b = blockIdx.z;
    const int w = threadIdx.x >> 5, lane = threadIdx.x & 31;
    const int wm = w >> 1, wn = w & 1;
    const int l0 = lane & 15, kh = lane >> 4;
    const int qbase = blockIdx.y * 64 + wm * 16;
    const int ibase = blockIdx.x * 128 + wn * 64;

    const bf16_t* sb = scoresB + (size_t)b * HH * HH;
    const bf16_t* gb = gB + (size_t)b * HH * ICC;
    const float* cm = colmax + b * HH;
    const float* ci = colinv + b * HH;

    v8f acc[4] = {};
    const bf16_t* srow = sb + (size_t)(qbase + l0) * HH;
    for (int kb = 0; kb < HH; kb += 32) {
        __builtin_prefetch(srow + kb + 512, 0, 0);   // stream next scores chunk
        v16bf a;
        #pragma unroll
        for (int j = 0; j < 16; ++j) {
            const int kk = kb + kh * 16 + j;
            const float sval = (float)srow[kk];
            a[j] = (bf16_t)(__expf(sval - cm[kk]) * ci[kk]);
        }
        #pragma unroll
        for (int t = 0; t < 4; ++t) {
            v16bf bfrag;
            const bf16_t* bcol = gb + (size_t)(kb + kh * 16) * ICC + ibase + t * 16 + l0;
            #pragma unroll
            for (int j = 0; j < 16; ++j) bfrag[j] = bcol[(size_t)j * ICC];
            acc[t] = __builtin_amdgcn_wmma_f32_16x16x32_bf16(
                false, a, false, bfrag, (short)0, acc[t], false, false);
        }
    }
    bf16_t* ab = agB + (size_t)b * HH * ICC;
    #pragma unroll
    for (int t = 0; t < 4; ++t)
        #pragma unroll
        for (int r = 0; r < 8; ++r) {
            const int q = qbase + kh * 8 + r;
            const int i = ibase + t * 16 + l0;
            ab[(size_t)q * ICC + i] = (bf16_t)acc[t][r];
        }
}

// -------------------------------------------------------------------------
// K5: out[c][h] = sum_i w_mask[c][i]*ag[h][i] + b_mask[c] + x[c][h]
//     (M=512, N=4096, K=256). Wave: M16 x N64. Block: 8 waves = M32 x N256.
// -------------------------------------------------------------------------
__global__ __launch_bounds__(256) void mask_kernel(
    const float* __restrict__ w_mask, const float* __restrict__ b_mask,
    const bf16_t* __restrict__ agB, const float* __restrict__ x,
    float* __restrict__ out)
{
    const int b = blockIdx.z;
    const int w = threadIdx.x >> 5, lane = threadIdx.x & 31;
    const int wm = w >> 2, wn = w & 3;
    const int l0 = lane & 15, kh = lane >> 4;
    const int cbase = blockIdx.y * 32 + wm * 16;
    const int hbase = blockIdx.x * 256 + wn * 64;

    const bf16_t* ab = agB + (size_t)b * HH * ICC;

    v8f acc[4] = {};
    for (int kb = 0; kb < ICC; kb += 32) {
        v16bf a;
        const float* wrow = w_mask + (size_t)(cbase + l0) * ICC + kb;
        #pragma unroll
        for (int j = 0; j < 16; ++j) a[j] = (bf16_t)wrow[kh * 16 + j];
        #pragma unroll
        for (int t = 0; t < 4; ++t) {
            v16bf bfrag;
            const bf16_t* brow = ab + (size_t)(hbase + t * 16 + l0) * ICC + kb + kh * 16;
            #pragma unroll
            for (int j = 0; j < 16; ++j) bfrag[j] = brow[j];
            acc[t] = __builtin_amdgcn_wmma_f32_16x16x32_bf16(
                false, a, false, bfrag, (short)0, acc[t], false, false);
        }
    }
    const float* xb = x + (size_t)b * CC * HH;
    float* ob = out + (size_t)b * CC * HH;
    #pragma unroll
    for (int t = 0; t < 4; ++t)
        #pragma unroll
        for (int r = 0; r < 8; ++r) {
            const int c = cbase + kh * 8 + r;
            const int h = hbase + t * 16 + l0;
            const size_t idx = (size_t)c * HH + h;
            ob[idx] = acc[t][r] + b_mask[c] + xb[idx];
        }
}

// -------------------------------------------------------------------------
extern "C" void kernel_launch(void* const* d_in, const int* in_sizes, int n_in,
                              void* d_out, int out_size, void* d_ws, size_t ws_size,
                              hipStream_t stream) {
    const float* x       = (const float*)d_in[0];
    const float* w_phi   = (const float*)d_in[1];
    const float* b_phi   = (const float*)d_in[2];
    const float* w_theta = (const float*)d_in[3];
    const float* b_theta = (const float*)d_in[4];
    const float* w_g     = (const float*)d_in[5];
    const float* b_g     = (const float*)d_in[6];
    const float* w_mask  = (const float*)d_in[7];
    const float* b_mask  = (const float*)d_in[8];
    float* out = (float*)d_out;

    char* ws = (char*)d_ws;
    size_t off = 0;
    bf16_t* phiB    = (bf16_t*)(ws + off); off += (size_t)BB * ICC * HH * sizeof(bf16_t);
    bf16_t* thetaB  = (bf16_t*)(ws + off); off += (size_t)BB * HH * ICC * sizeof(bf16_t);
    bf16_t* gB      = (bf16_t*)(ws + off); off += (size_t)BB * HH * ICC * sizeof(bf16_t);
    bf16_t* agB     = (bf16_t*)(ws + off); off += (size_t)BB * HH * ICC * sizeof(bf16_t);
    float*  colmax  = (float*)(ws + off);  off += (size_t)BB * HH * sizeof(float);
    float*  colinv  = (float*)(ws + off);  off += (size_t)BB * HH * sizeof(float);
    bf16_t* scoresB = (bf16_t*)(ws + off); off += (size_t)BB * HH * HH * sizeof(bf16_t);

    proj_kernel<<<dim3(HH / 256, ICC / 32, BB * 3), 256, 0, stream>>>(
        x, w_phi, b_phi, w_theta, b_theta, w_g, b_g, phiB, thetaB, gB);

    scores_kernel<<<dim3(HH / 128, HH / 64, BB), 256, 0, stream>>>(
        thetaB, phiB, scoresB);

    colsoftmax_kernel<<<dim3(HH / 256, BB), 256, 0, stream>>>(
        scoresB, colmax, colinv);

    ag_kernel<<<dim3(ICC / 128, HH / 64, BB), 256, 0, stream>>>(
        scoresB, colmax, colinv, gB, agB);

    mask_kernel<<<dim3(HH / 256, CC / 32, BB), 256, 0, stream>>>(
        w_mask, b_mask, agB, x, out);
}
